// RAPMG_79542794322036
// MI455X (gfx1250) — matrix operand order repaired and checked by reading, not compiled, not run on hardware
//
#include <hip/hip_runtime.h>
#include <hip/hip_bf16.h>
#include <float.h>

// ---------------- constants (from reference) ----------------
#define BATCH 32
#define CH    512
#define HF    32
#define IMGSZ 256
#define NTOK  256      // PATCH*PATCH
#define HEADS 8
#define HD    64       // head dim
#define QKVD  1536     // 3*CH
#define WIN   230
#define REMAIN 115

typedef __attribute__((ext_vector_type(16))) __bf16 v16bf;
typedef __attribute__((ext_vector_type(8)))  float  v8f;

struct __attribute__((aligned(16))) U4 { unsigned int x, y, z, w; };
union FragU { U4 q[2]; v16bf v; };

#if defined(__has_builtin)
#if __has_builtin(__builtin_amdgcn_global_load_async_to_lds_b128) && \
    __has_builtin(__builtin_amdgcn_s_wait_asynccnt)
#define HAVE_ASYNC_LDS 1
#endif
#endif

typedef int vi4 __attribute__((vector_size(16)));

__device__ __forceinline__ void cp16_to_lds(void* l, const void* g) {
#ifdef HAVE_ASYNC_LDS
    __builtin_amdgcn_global_load_async_to_lds_b128(
        (__attribute__((address_space(1))) vi4*)g,
        (__attribute__((address_space(3))) vi4*)l, 0, 0);
#else
    *(U4*)l = *(const U4*)g;
#endif
}
__device__ __forceinline__ void cp_lds_wait() {
#ifdef HAVE_ASYNC_LDS
    __builtin_amdgcn_s_wait_asynccnt(0);
#endif
}

__device__ __forceinline__ __bf16 f2bf(float f) {
    union { float f; unsigned u; } a; a.f = f;
    unsigned r = a.u + 0x7FFFu + ((a.u >> 16) & 1u);   // RNE
    union { unsigned short s; __bf16 b; } o; o.s = (unsigned short)(r >> 16);
    return o.b;
}
__device__ __forceinline__ float bf2f(__bf16 b) {
    union { unsigned short s; __bf16 b; } i; i.b = b;
    union { unsigned u; float f; } o; o.u = ((unsigned)i.s) << 16;
    return o.f;
}

// A-matrix fragment (16x32 bf16, row-major source, lda in elements)
// ISA layout: lane(0-15)->M, elements e: K = (e/8)*16 + (lane/16)*8 + e%8
__device__ __forceinline__ v16bf load_fragA(const __bf16* base, int lda,
                                            int row0, int k0, int lane) {
    int m = lane & 15, half = lane >> 4;
    const __bf16* p = base + (size_t)(row0 + m) * lda + k0;
    FragU u;
    u.q[0] = *(const U4*)(p + half * 8);        // K = half*8 .. +7
    u.q[1] = *(const U4*)(p + 16 + half * 8);   // K = 16+half*8 .. +7
    return u.v;
}
// B-matrix fragment (32x16 bf16). Source row-major [N][K] (B[k][n]=W[n][k]).
// ISA layout: lane(0-15)->N, elements e: K = (e/8)*16 + (lane/16)*8 + e%8 (matches fragA)
__device__ __forceinline__ v16bf load_fragB(const __bf16* base, int ldb,
                                            int col0, int k0, int lane) {
    int n = lane & 15, half = lane >> 4;
    const __bf16* p = base + (size_t)(col0 + n) * ldb + k0 + half * 8;
    FragU u;
    u.q[0] = *(const U4*)(p);
    u.q[1] = *(const U4*)(p + 16);
    return u.v;
}

// ---------------- K0: fp32 -> bf16 weight convert ----------------
__global__ void k_cvt_bf16(const float* __restrict__ in, __bf16* __restrict__ out, int n) {
    int i = blockIdx.x * 256 + threadIdx.x;
    if (i < n) out[i] = f2bf(in[i]);
}

// ---------------- K1: per-(b,c) mean/max over 32x32 ----------------
__global__ void k_chan_reduce(const float* __restrict__ xf3,
                              float* __restrict__ vavg, float* __restrict__ vmax) {
    int bc = blockIdx.x;                       // B*C blocks
    const float* p = xf3 + (size_t)bc * 1024;
    int tid = threadIdx.x;
    float s = 0.f, m = -FLT_MAX;
    for (int i = tid; i < 1024; i += 256) { float v = p[i]; s += v; m = fmaxf(m, v); }
    __shared__ float ss[256], sm[256];
    ss[tid] = s; sm[tid] = m; __syncthreads();
    for (int o = 128; o > 0; o >>= 1) {
        if (tid < o) { ss[tid] += ss[tid + o]; sm[tid] = fmaxf(sm[tid], sm[tid + o]); }
        __syncthreads();
    }
    if (tid == 0) { vavg[bc] = ss[0] * (1.f / 1024.f); vmax[bc] = sm[0]; }
}

// ---------------- K2: channel-attention MLP + sigmoid ----------------
__global__ void k_ca_mlp(const float* __restrict__ vavg, const float* __restrict__ vmax,
                         const float* __restrict__ fc1,  const float* __restrict__ fc2,
                         float* __restrict__ ca) {
    int b = blockIdx.x, tid = threadIdx.x;
    __shared__ float hsum[32];
    const float* va = vavg + b * CH;
    const float* vm = vmax + b * CH;
    if (tid < 32) {
        float sa = 0.f, sm = 0.f;
        const float* w = fc1 + tid * CH;       // fc1: (32, 512)
        for (int c = 0; c < CH; ++c) { float wv = w[c]; sa += wv * va[c]; sm += wv * vm[c]; }
        hsum[tid] = fmaxf(sa, 0.f) + fmaxf(sm, 0.f);
    }
    __syncthreads();
    for (int c = tid; c < CH; c += 256) {
        float s = 0.f;
        const float* w2 = fc2 + c * 32;        // fc2: (512, 32)
        for (int j = 0; j < 32; ++j) s += w2[j] * hsum[j];
        ca[b * CH + c] = 1.f / (1.f + __expf(-s));
    }
}

// ---------------- K3: spatial stats (mean/max over C of x*ca) ----------------
__global__ void k_spatial_stats(const float* __restrict__ xf3, const float* __restrict__ ca,
                                float* __restrict__ s_in) {
    int b = blockIdx.x, pix = threadIdx.x;     // 1024 threads
    const float* xb = xf3 + (size_t)b * CH * 1024 + pix;
    const float* cb = ca + b * CH;
    float s = 0.f, m = -FLT_MAX;
    for (int c = 0; c < CH; ++c) { float v = xb[(size_t)c * 1024] * cb[c]; s += v; m = fmaxf(m, v); }
    s_in[b * 2048 + pix]        = s * (1.f / (float)CH);
    s_in[b * 2048 + 1024 + pix] = m;
}

// ---------------- K4: 7x7 conv (pad 3) + sigmoid ----------------
__global__ void k_sconv(const float* __restrict__ s_in, const float* __restrict__ w,
                        float* __restrict__ sgate) {
    int b = blockIdx.x, pix = threadIdx.x;     // 1024 threads
    int y = pix >> 5, x = pix & 31;
    __shared__ float sw[98];
    __shared__ float tile[2048];
    if (pix < 98) sw[pix] = w[pix];
    tile[pix]        = s_in[b * 2048 + pix];
    tile[pix + 1024] = s_in[b * 2048 + 1024 + pix];
    __syncthreads();
    float acc = 0.f;
    for (int c = 0; c < 2; ++c)
        for (int ky = 0; ky < 7; ++ky) {
            int yy = y + ky - 3; if (yy < 0 || yy > 31) continue;
            for (int kx = 0; kx < 7; ++kx) {
                int xx = x + kx - 3; if (xx < 0 || xx > 31) continue;
                acc += tile[c * 1024 + yy * 32 + xx] * sw[c * 49 + ky * 7 + kx];
            }
        }
    sgate[b * 1024 + pix] = 1.f / (1.f + __expf(-acc));
}

// ---------------- K5: 2x2 pooled tokens + LayerNorm -> bf16 ----------------
__global__ void k_tokens_ln(const float* __restrict__ xf3, const float* __restrict__ ca,
                            const float* __restrict__ sgate,
                            const float* __restrict__ ln_g, const float* __restrict__ ln_b,
                            __bf16* __restrict__ tn) {
    int bn = blockIdx.x;                        // B*NTOK blocks
    int b = bn >> 8, n = bn & 255;
    int py = n >> 4, px = n & 15;
    int bp = (py * 2) * 32 + px * 2;
    float g00 = sgate[b * 1024 + bp],      g01 = sgate[b * 1024 + bp + 1];
    float g10 = sgate[b * 1024 + bp + 32], g11 = sgate[b * 1024 + bp + 33];
    int tid = threadIdx.x;
    float vals[2];
    for (int i = 0; i < 2; ++i) {
        int c = tid + i * 256;
        const float* xp = xf3 + ((size_t)b * CH + c) * 1024 + bp;
        float cc = ca[b * CH + c];
        vals[i] = (xp[0] * g00 + xp[1] * g01 + xp[32] * g10 + xp[33] * g11) * cc * 0.25f;
    }
    __shared__ float rs[256], rq[256];
    rs[tid] = vals[0] + vals[1];
    rq[tid] = vals[0] * vals[0] + vals[1] * vals[1];
    __syncthreads();
    for (int o = 128; o > 0; o >>= 1) {
        if (tid < o) { rs[tid] += rs[tid + o]; rq[tid] += rq[tid + o]; }
        __syncthreads();
    }
    __shared__ float smu, sinv;
    if (tid == 0) {
        float mu = rs[0] * (1.f / (float)CH);
        float var = rq[0] * (1.f / (float)CH) - mu * mu;
        smu = mu; sinv = rsqrtf(var + 1e-5f);
    }
    __syncthreads();
    float mu = smu, inv = sinv;
    __bf16* out = tn + ((size_t)b * NTOK + n) * CH;
    for (int i = 0; i < 2; ++i) {
        int c = tid + i * 256;
        out[c] = f2bf((vals[i] - mu) * inv * ln_g[c] + ln_b[c]);
    }
}

// ---------------- K6: QKV GEMM (WMMA bf16, A staged in LDS) ----------------
// block = 8 waves; block owns one 16-row A tile (16x512, LDS) and 512 output cols;
// each wave computes a 16x64 strip (4 accumulators, A-fragment reused 4x).
__global__ void k_qkv_gemm(const __bf16* __restrict__ tn, const __bf16* __restrict__ wq,
                           const float* __restrict__ qkv_b, __bf16* __restrict__ qkv) {
    __shared__ __align__(16) __bf16 sA[16 * CH];   // 16KB
    int blk = blockIdx.x;                          // BATCH*16*3 blocks
    int b = blk / 48, r = blk % 48;
    int mt = r / 3, ng = r % 3;
    int tid = threadIdx.x;
    const __bf16* Ag = tn + ((size_t)b * NTOK + mt * 16) * CH;
    for (int i = tid; i < 1024; i += 256)          // 1024 x 16B
        cp16_to_lds((char*)sA + (size_t)i * 16, (const char*)Ag + (size_t)i * 16);
    cp_lds_wait();
    __syncthreads();

    int wv = tid >> 5, lane = tid & 31;
    int col0 = ng * 512 + wv * 64;                 // wave's 64 output columns
    v8f acc[4] = {};
    for (int ks = 0; ks < 16; ++ks) {
        v16bf a = load_fragA(sA, CH, 0, ks * 32, lane);   // ds_load_b128 x2
#pragma unroll
        for (int j = 0; j < 4; ++j) {
            if (ks < 15)
                __builtin_prefetch(wq + (size_t)(col0 + j * 16 + (lane & 15)) * CH + (ks + 1) * 32, 0, 3);
            v16bf bb = load_fragB(wq, CH, col0 + j * 16, ks * 32, lane);
            acc[j] = __builtin_amdgcn_wmma_f32_16x16x32_bf16(false, a, false, bb,
                                                             (short)0, acc[j], false, false);
        }
    }
    int half = lane >> 4;
    __bf16* outp = qkv + (size_t)b * NTOK * QKVD;
#pragma unroll
    for (int j = 0; j < 4; ++j) {
        int col = col0 + j * 16 + (lane & 15);
        float bias = qkv_b[col];
#pragma unroll
        for (int v = 0; v < 8; ++v)
            outp[(size_t)(mt * 16 + v + 8 * half) * QKVD + col] = f2bf(acc[j][v] + bias);
    }
}

// ---------------- K7: attention scores S = (q k^T) * scale  (WMMA) ----------------
// wave computes 16x64 block; A fragments (K=64) kept in registers across 4 N-tiles
__global__ void k_attn_scores(const __bf16* __restrict__ qkv, float* __restrict__ scores) {
    int wid = blockIdx.x * 8 + (threadIdx.x >> 5);   // 256bh * 16mt * 4ng waves
    int lane = threadIdx.x & 31;
    int bh = wid >> 6;
    int t = wid & 63;
    int mt = t >> 2, ng = t & 3;
    int b = bh >> 3, h = bh & 7;
    const __bf16* Ab = qkv + (size_t)b * NTOK * QKVD + h * HD;        // q
    const __bf16* Bb = qkv + (size_t)b * NTOK * QKVD + CH + h * HD;   // k
    v16bf a0 = load_fragA(Ab, QKVD, mt * 16, 0, lane);
    v16bf a1 = load_fragA(Ab, QKVD, mt * 16, 32, lane);
    float* out = scores + (size_t)bh * 65536;
    int half = lane >> 4;
#pragma unroll
    for (int j = 0; j < 4; ++j) {
        int nt = ng * 4 + j;
        v16bf b0 = load_fragB(Bb, QKVD, nt * 16, 0, lane);
        v16bf b1 = load_fragB(Bb, QKVD, nt * 16, 32, lane);
        v8f acc = {};
        acc = __builtin_amdgcn_wmma_f32_16x16x32_bf16(false, a0, false, b0,
                                                      (short)0, acc, false, false);
        acc = __builtin_amdgcn_wmma_f32_16x16x32_bf16(false, a1, false, b1,
                                                      (short)0, acc, false, false);
        int col = nt * 16 + (lane & 15);
#pragma unroll
        for (int v = 0; v < 8; ++v)
            out[(size_t)(mt * 16 + v + 8 * half) * 256 + col] = acc[v] * 0.125f;  // 64^-0.5
    }
}

// ---------------- K8: row softmax -> bf16 probs ----------------
__global__ void k_softmax(const float* __restrict__ scores, __bf16* __restrict__ probs) {
    int row = blockIdx.x;              // 32*8*256 rows
    int tid = threadIdx.x;             // 256 threads
    float v = scores[(size_t)row * 256 + tid];
    __shared__ float red[256];
    __shared__ float bc;
    red[tid] = v; __syncthreads();
    for (int o = 128; o > 0; o >>= 1) {
        if (tid < o) red[tid] = fmaxf(red[tid], red[tid + o]);
        __syncthreads();
    }
    if (tid == 0) bc = red[0];
    __syncthreads();
    float e = __expf(v - bc);
    __syncthreads();
    red[tid] = e; __syncthreads();
    for (int o = 128; o > 0; o >>= 1) {
        if (tid < o) red[tid] += red[tid + o];
        __syncthreads();
    }
    if (tid == 0) bc = red[0];
    __syncthreads();
    probs[(size_t)row * 256 + tid] = f2bf(e / bc);
}

// ---------------- K8b: transpose V head-slices -> vT[bh][d][m] ----------------
// gives the AV GEMM a row-major [N=d][K=m] B-source with contiguous-K b128 loads
__global__ void k_vT(const __bf16* __restrict__ qkv, __bf16* __restrict__ vT) {
    int bh = blockIdx.x;               // 256 blocks
    int b = bh >> 3, h = bh & 7;
    int m = threadIdx.x;               // 256 threads
    const __bf16* src = qkv + (size_t)b * NTOK * QKVD + (size_t)m * QKVD + 2 * CH + h * HD;
    union { U4 q[8]; __bf16 s[64]; } u;
#pragma unroll
    for (int j = 0; j < 8; ++j) u.q[j] = ((const U4*)src)[j];
    __bf16* dst = vT + (size_t)bh * HD * NTOK + m;
#pragma unroll
    for (int d = 0; d < 64; ++d) dst[(size_t)d * NTOK] = u.s[d];   // coalesced across threads
}

// ---------------- K9: O = P V  (WMMA) ----------------
// wave computes a full 16(row) x 64(head-dim) block; A reused across 4 d-tiles
__global__ void k_attn_av(const __bf16* __restrict__ probs, const __bf16* __restrict__ vT,
                          __bf16* __restrict__ attno) {
    int wid = blockIdx.x * 8 + (threadIdx.x >> 5);   // 256bh * 16mt waves
    int lane = threadIdx.x & 31;
    int bh = wid >> 4;
    int mt = wid & 15;
    int b = bh >> 3, h = bh & 7;
    const __bf16* A = probs + (size_t)bh * 65536;
    const __bf16* Bb = vT + (size_t)bh * HD * NTOK;   // [d][m], ld = 256
    v8f acc[4] = {};
    for (int ks = 0; ks < 8; ++ks) {
        if (ks < 7)
            __builtin_prefetch(A + (size_t)(mt * 16 + (lane & 15)) * 256 + (ks + 1) * 32, 0, 3);
        v16bf a = load_fragA(A, 256, mt * 16, ks * 32, lane);
#pragma unroll
        for (int j = 0; j < 4; ++j) {
            v16bf bb = load_fragB(Bb, NTOK, j * 16, ks * 32, lane);
            acc[j] = __builtin_amdgcn_wmma_f32_16x16x32_bf16(false, a, false, bb,
                                                             (short)0, acc[j], false, false);
        }
    }
    __bf16* out = attno + (size_t)b * NTOK * CH;
    int half = lane >> 4;
#pragma unroll
    for (int j = 0; j < 4; ++j) {
        int col = h * HD + j * 16 + (lane & 15);
#pragma unroll
        for (int v = 0; v < 8; ++v)
            out[(size_t)(mt * 16 + v + 8 * half) * CH + col] = f2bf(acc[j][v]);
    }
}

// ---------------- K10: mean over tokens ----------------
__global__ void k_attn_mean(const __bf16* __restrict__ attno, float* __restrict__ omean) {
    int b = blockIdx.x, c = threadIdx.x;   // 512 threads
    const __bf16* p = attno + (size_t)b * NTOK * CH + c;
    float s = 0.f;
    for (int n = 0; n < NTOK; ++n) s += bf2f(p[(size_t)n * CH]);
    omean[b * CH + c] = s * (1.f / (float)NTOK);
}

// ---------------- K11: proj matvec + classifier head ----------------
__global__ void k_proj_fc(const float* __restrict__ omean,
                          const float* __restrict__ proj_w, const float* __restrict__ proj_b,
                          const float* __restrict__ fcw_w,  const float* __restrict__ fcw_b,
                          float* __restrict__ out1) {
    int b = blockIdx.x, c = threadIdx.x;   // 512 threads
    __shared__ float vsh[512], feat[512];
    vsh[c] = omean[b * CH + c];
    __syncthreads();
    float s = 0.f;
    const float* w = proj_w + c * CH;      // proj_w: (512, 512)
    for (int j = 0; j < CH; ++j) s += w[j] * vsh[j];
    feat[c] = s + proj_b[c];
    __syncthreads();
    if (c < 4) {
        float o = 0.f;
        const float* fw = fcw_w + c * CH;  // fcw_w: (4, 512)
        for (int j = 0; j < CH; ++j) o += fw[j] * feat[j];
        out1[b * 4 + c] = o + fcw_b[c];
    }
}

// ---------------- K12: CAM + first-index argmax -> crop origin ----------------
__global__ void k_cam_argmax(const float* __restrict__ xf3, const float* __restrict__ vavg,
                             int* __restrict__ axay) {
    int b = blockIdx.x, tid = threadIdx.x;   // 256 threads
    __shared__ float vsh[512];
    vsh[tid] = vavg[b * CH + tid];
    vsh[tid + 256] = vavg[b * CH + tid + 256];
    __syncthreads();
    float best = -FLT_MAX; int bidx = 0x7FFFFFFF;
    for (int i = 0; i < 4; ++i) {
        int pix = tid + i * 256;
        const float* xp = xf3 + (size_t)b * CH * 1024 + pix;
        float s = 0.f;
        for (int c = 0; c < CH; ++c) s += fmaxf(vsh[c] * xp[(size_t)c * 1024], 0.f);
        if (s > best || (s == best && pix < bidx)) { best = s; bidx = pix; }
    }
    __shared__ float bv[256]; __shared__ int bi[256];
    bv[tid] = best; bi[tid] = bidx; __syncthreads();
    for (int o = 128; o > 0; o >>= 1) {
        if (tid < o) {
            if (bv[tid + o] > bv[tid] || (bv[tid + o] == bv[tid] && bi[tid + o] < bi[tid])) {
                bv[tid] = bv[tid + o]; bi[tid] = bi[tid + o];
            }
        }
        __syncthreads();
    }
    if (tid == 0) {
        int idx = bi[0];
        int fx = idx >> 5, fy = idx & 31;
        int ix = 8 * fx, iy = 8 * fy;
        int ax = min(max(ix - REMAIN, 0), IMGSZ - WIN);
        int ay = min(max(iy - REMAIN, 0), IMGSZ - WIN);
        axay[b * 2] = ax; axay[b * 2 + 1] = ay;
    }
}

// ---------------- K13: crop + align-corners bilinear resize ----------------
__global__ void k_crop_resize(const float* __restrict__ x, const int* __restrict__ axay,
                              float* __restrict__ out) {
    int gid = blockIdx.x;                  // B*3*256 blocks (one output row each)
    int b = gid / 768, r = gid % 768;
    int c = r / 256, y = r % 256;
    int ax = axay[b * 2], ay = axay[b * 2 + 1];
    const float* img = x + ((size_t)b * 3 + c) * (IMGSZ * IMGSZ);
    const float scl = (float)(WIN - 1) / (float)(IMGSZ - 1);
    float cy = y * scl;
    int y0 = min(max((int)floorf(cy), 0), WIN - 1);
    int y1 = min(y0 + 1, WIN - 1);
    float fy_ = cy - (float)y0;
    int xo = threadIdx.x;
    float cx = xo * scl;
    int x0 = min(max((int)floorf(cx), 0), WIN - 1);
    int x1 = min(x0 + 1, WIN - 1);
    float fx_ = cx - (float)x0;
    const float* r0 = img + (size_t)(ax + y0) * IMGSZ + ay;
    const float* r1 = img + (size_t)(ax + y1) * IMGSZ + ay;
    float top = r0[x0] * (1.f - fx_) + r0[x1] * fx_;
    float bot = r1[x0] * (1.f - fx_) + r1[x1] * fx_;
    out[128 + ((size_t)b * 3 + c) * (IMGSZ * IMGSZ) + (size_t)y * IMGSZ + xo] =
        top * (1.f - fy_) + bot * fy_;
}

// ---------------- launch ----------------
extern "C" void kernel_launch(void* const* d_in, const int* in_sizes, int n_in,
                              void* d_out, int out_size, void* d_ws, size_t ws_size,
                              hipStream_t stream) {
    const float* x       = (const float*)d_in[0];
    const float* xf3     = (const float*)d_in[1];
    const float* fc1     = (const float*)d_in[2];
    const float* fc2     = (const float*)d_in[3];
    const float* sconv_w = (const float*)d_in[4];
    const float* ln_g    = (const float*)d_in[5];
    const float* ln_b    = (const float*)d_in[6];
    const float* qkv_w   = (const float*)d_in[7];
    const float* qkv_b   = (const float*)d_in[8];
    const float* proj_w  = (const float*)d_in[9];
    const float* proj_b  = (const float*)d_in[10];
    const float* fcw_w   = (const float*)d_in[11];
    const float* fcw_b   = (const float*)d_in[12];
    float* out = (float*)d_out;

    char* base = (char*)d_ws;
    size_t off = 0;
    auto alloc = [&](size_t bytes) -> char* {
        char* p = base + off;
        off += (bytes + 255) & ~(size_t)255;
        return p;
    };
    float*  vavg    = (float*)alloc((size_t)BATCH * CH * 4);
    float*  vmax    = (float*)alloc((size_t)BATCH * CH * 4);
    float*  ca      = (float*)alloc((size_t)BATCH * CH * 4);
    float*  s_in    = (float*)alloc((size_t)BATCH * 2048 * 4);
    float*  sgate   = (float*)alloc((size_t)BATCH * 1024 * 4);
    int*    axay    = (int*)  alloc((size_t)BATCH * 2 * 4);
    __bf16* tn      = (__bf16*)alloc((size_t)BATCH * NTOK * CH * 2);
    __bf16* wq_bf   = (__bf16*)alloc((size_t)QKVD * CH * 2);
    __bf16* qkv     = (__bf16*)alloc((size_t)BATCH * NTOK * QKVD * 2);
    float*  scores  = (float*)alloc((size_t)BATCH * HEADS * 65536 * 4);
    __bf16* probs   = (__bf16*)alloc((size_t)BATCH * HEADS * 65536 * 2);
    __bf16* vT      = (__bf16*)alloc((size_t)BATCH * HEADS * HD * NTOK * 2);
    __bf16* attno   = (__bf16*)alloc((size_t)BATCH * NTOK * CH * 2);
    float*  omean   = (float*)alloc((size_t)BATCH * CH * 4);
    (void)ws_size; (void)n_in; (void)in_sizes; (void)out_size;

    // weights -> bf16
    k_cvt_bf16<<<(QKVD * CH + 255) / 256, 256, 0, stream>>>(qkv_w, wq_bf, QKVD * CH);
    // CBAM
    k_chan_reduce<<<BATCH * CH, 256, 0, stream>>>(xf3, vavg, vmax);
    k_ca_mlp<<<BATCH, 256, 0, stream>>>(vavg, vmax, fc1, fc2, ca);
    k_spatial_stats<<<BATCH, 1024, 0, stream>>>(xf3, ca, s_in);
    k_sconv<<<BATCH, 1024, 0, stream>>>(s_in, sconv_w, sgate);
    // tokens + LN
    k_tokens_ln<<<BATCH * NTOK, 256, 0, stream>>>(xf3, ca, sgate, ln_g, ln_b, tn);
    // attention (WMMA bf16)
    k_qkv_gemm<<<BATCH * 16 * 3, 256, 0, stream>>>(tn, wq_bf, qkv_b, qkv);
    k_attn_scores<<<(BATCH * HEADS * 16 * 4) / 8, 256, 0, stream>>>(qkv, scores);
    k_softmax<<<BATCH * HEADS * 256, 256, 0, stream>>>(scores, probs);
    k_vT<<<BATCH * HEADS, 256, 0, stream>>>(qkv, vT);
    k_attn_av<<<(BATCH * HEADS * 16) / 8, 256, 0, stream>>>(probs, vT, attno);
    k_attn_mean<<<BATCH, 512, 0, stream>>>(attno, omean);
    k_proj_fc<<<BATCH, 512, 0, stream>>>(omean, proj_w, proj_b, fcw_w, fcw_b, out);
    // image cut branch
    k_cam_argmax<<<BATCH, 256, 0, stream>>>(xf3, vavg, axay);
    k_crop_resize<<<BATCH * 3 * 256, 256, 0, stream>>>(x, axay, out);
}
